// NeuralNet_27943057228550
// MI455X (gfx1250) — matrix operand order repaired and checked by reading, not compiled, not run on hardware
//
#include <hip/hip_runtime.h>
#include <hip/hip_bf16.h>

// ---------------------------------------------------------------------------
// CDNA5 (gfx1250) bf16-WMMA MLP forward.
// GEMMs run as D = A(bf16) x W^T(bf16) + bias with f32 accumulation using
// v_wmma_f32_16x16x32_bf16.  Tile staging uses the gfx1250 async
// global->LDS path (ASYNCcnt) when the builtin is available.
// ---------------------------------------------------------------------------

typedef __attribute__((ext_vector_type(16))) __bf16 v16bf;
typedef __attribute__((ext_vector_type(8)))  __bf16 bf16x8;
typedef __attribute__((ext_vector_type(8)))  float  v8f;
typedef __attribute__((ext_vector_type(4)))  int    v4i;

#define AS1 __attribute__((address_space(1)))
#define AS3 __attribute__((address_space(3)))

#define LRELU_SLOPE 0.01f
#define KEEP_SCALE  2.0f

#ifdef __has_builtin
#  if __has_builtin(__builtin_amdgcn_global_load_async_to_lds_b128)
#    define HAVE_ASYNC_LDS 1
#  endif
#endif
#ifndef HAVE_ASYNC_LDS
#  define HAVE_ASYNC_LDS 0
#endif

#define MT 128   // rows per workgroup (8 waves x 16)
#define NT 64    // cols per workgroup (4 x 16 per wave)
#define KT 32    // k-step (bf16 WMMA K)
#define LDS_STRIDE 40  // padded row stride (bf16 elems) to spread LDS banks

__device__ __forceinline__ void async_copy_b128(const __bf16* g, __bf16* l) {
#if HAVE_ASYNC_LDS
    __bf16* gnc = const_cast<__bf16*>(g);
    __builtin_amdgcn_global_load_async_to_lds_b128(
        (AS1 v4i*)gnc, (AS3 v4i*)l, 0, 0);
#else
    *(bf16x8*)l = *(const bf16x8*)g;
#endif
}

__global__ __launch_bounds__(256) void gemm_bf16_wmma(
    const __bf16* __restrict__ A,    // [M, K] row-major activations
    const __bf16* __restrict__ W,    // [N, K] row-major weights (row n = out col n)
    const float*  __restrict__ bias, // [N]
    float*        __restrict__ Z,    // [M, N] row-major pre-activations
    int K, int Nstride)
{
    __shared__ __bf16 lA[2][MT * LDS_STRIDE];
    __shared__ __bf16 lB[2][NT * LDS_STRIDE];

    const int row0 = blockIdx.y * MT;
    const int col0 = blockIdx.x * NT;
    const int tid  = threadIdx.x;
    const int wv   = tid >> 5;
    const int lane = tid & 31;

    auto issue_tile = [&](int buf, int kk) {
        // A tile: 128x32 bf16 = 512 x 16B chunks, 2 per thread
        #pragma unroll
        for (int i = 0; i < 2; ++i) {
            int c   = tid + i * 256;
            int r   = c >> 2;
            int col = (c & 3) << 3;
            async_copy_b128(A + (size_t)(row0 + r) * K + kk + col,
                            &lA[buf][r * LDS_STRIDE + col]);
        }
        // B tile: 64x32 bf16 = 256 x 16B chunks, 1 per thread
        {
            int r   = tid >> 2;
            int col = (tid & 3) << 3;
            async_copy_b128(W + (size_t)(col0 + r) * K + kk + col,
                            &lB[buf][r * LDS_STRIDE + col]);
        }
    };

    auto wait_tiles = [&]() {
#if HAVE_ASYNC_LDS
#  if __has_builtin(__builtin_amdgcn_s_wait_asynccnt)
        __builtin_amdgcn_s_wait_asynccnt(0);
#  else
        asm volatile("s_wait_asynccnt 0" ::: "memory");
#  endif
#endif
        __syncthreads();
    };

    // Accumulators: 4 n-subtiles of 16x16; fold bias in (N = lane%16 per tile).
    v8f acc[4];
    #pragma unroll
    for (int nt = 0; nt < 4; ++nt) {
        float bv = bias[col0 + nt * 16 + (lane & 15)];
        #pragma unroll
        for (int j = 0; j < 8; ++j) acc[nt][j] = bv;
    }

    issue_tile(0, 0);
    wait_tiles();

    // ISA fragment addressing (wave32):
    //  A 16x32 bf16: lane holds row m=lane%16, K runs [k0..k0+7] and [k0+16..k0+23], k0=8*(lane/16)
    //  B 32x16 bf16: lane holds col n=lane%16, K runs [kb..kb+7] and [kb+16..kb+23], kb=8*(lane/16)
    const int mbase = wv * 16 + (lane & 15);
    const int kA = (lane >> 4) << 3;
    const int kB = (lane >> 4) << 3;

    const int ksteps = K / KT;
    for (int s = 0; s < ksteps; ++s) {
        const int buf = s & 1;
        if (s + 1 < ksteps) issue_tile(buf ^ 1, (s + 1) * KT);

        union { v16bf v; bf16x8 h[2]; } ua;
        const __bf16* ap = &lA[buf][mbase * LDS_STRIDE + kA];
        ua.h[0] = *(const bf16x8*)ap;
        ua.h[1] = *(const bf16x8*)(ap + 16);
        const v16bf afrag = ua.v;

        #pragma unroll
        for (int nt = 0; nt < 4; ++nt) {
            union { v16bf v; bf16x8 h[2]; } ub;
            const __bf16* bp = &lB[buf][(nt * 16 + (lane & 15)) * LDS_STRIDE + kB];
            ub.h[0] = *(const bf16x8*)bp;
            ub.h[1] = *(const bf16x8*)(bp + 16);
            acc[nt] = __builtin_amdgcn_wmma_f32_16x16x32_bf16(
                false, afrag, false, ub.v, (short)0, acc[nt], false, false);
        }
        wait_tiles();
    }

    // C/D layout: VGPR v holds M = v + 8*(lane/16), N = lane%16.
    const int mrow = row0 + wv * 16 + ((lane >> 4) << 3);
    #pragma unroll
    for (int nt = 0; nt < 4; ++nt) {
        const int n = col0 + nt * 16 + (lane & 15);
        #pragma unroll
        for (int v = 0; v < 8; ++v) {
            Z[(size_t)(mrow + v) * Nstride + n] = acc[nt][v];
        }
    }
}

// ---------------------------------------------------------------------------
// Mixed activation + dropout-mask scale + bf16 cast. One block per row.
// tid: 0 relu, 1 tanh, 2 sigmoid, 3 softmax(type-3 cols), 4 softmax(-x, type-4
// cols), 5 exact gelu, 6 leaky relu.
// ---------------------------------------------------------------------------
__global__ __launch_bounds__(256) void act_mixed(
    const float* __restrict__ Z, const int* __restrict__ tids,
    const float* __restrict__ mask, __bf16* __restrict__ Hout, int H)
{
    __shared__ float red[4][8];
    const int r = blockIdx.x;
    const float* z  = Z    + (size_t)r * H;
    const float* mk = mask + (size_t)r * H;
    __bf16*      ho = Hout + (size_t)r * H;
    const int tid = threadIdx.x, wv = tid >> 5, lane = tid & 31;

    // Pass 1: restricted maxes for softmax(type3) and softmax(-x, type4)
    float m3 = -3.0e38f, m4 = -3.0e38f;
    for (int c = tid; c < H; c += 256) {
        float x = z[c]; int t = tids[c];
        if (t == 3) m3 = fmaxf(m3, x);
        if (t == 4) m4 = fmaxf(m4, -x);
    }
    #pragma unroll
    for (int off = 16; off > 0; off >>= 1) {
        m3 = fmaxf(m3, __shfl_xor(m3, off, 32));
        m4 = fmaxf(m4, __shfl_xor(m4, off, 32));
    }
    if (lane == 0) { red[0][wv] = m3; red[1][wv] = m4; }
    __syncthreads();
    m3 = red[0][0]; m4 = red[1][0];
    #pragma unroll
    for (int i = 1; i < 8; ++i) { m3 = fmaxf(m3, red[0][i]); m4 = fmaxf(m4, red[1][i]); }

    // Pass 2: restricted exp-sums
    float s3 = 0.f, s4 = 0.f;
    for (int c = tid; c < H; c += 256) {
        float x = z[c]; int t = tids[c];
        if (t == 3) s3 += expf(x - m3);
        if (t == 4) s4 += expf(-x - m4);
    }
    #pragma unroll
    for (int off = 16; off > 0; off >>= 1) {
        s3 += __shfl_xor(s3, off, 32);
        s4 += __shfl_xor(s4, off, 32);
    }
    if (lane == 0) { red[2][wv] = s3; red[3][wv] = s4; }
    __syncthreads();
    s3 = 0.f; s4 = 0.f;
    #pragma unroll
    for (int i = 0; i < 8; ++i) { s3 += red[2][i]; s4 += red[3][i]; }
    const float inv3 = 1.0f / s3, inv4 = 1.0f / s4;  // unused if set empty

    // Pass 3: apply activation, mask * 2, cast to bf16 for next GEMM
    for (int c = tid; c < H; c += 256) {
        float x = z[c]; int t = tids[c];
        float v;
        switch (t) {
            case 0:  v = fmaxf(x, 0.f); break;
            case 1:  v = tanhf(x); break;
            case 2:  v = 1.0f / (1.0f + expf(-x)); break;
            case 3:  v = expf(x - m3) * inv3; break;
            case 4:  v = expf(-x - m4) * inv4; break;
            case 5:  v = 0.5f * x * (1.0f + erff(x * 0.70710678f)); break;
            default: v = (x >= 0.f) ? x : LRELU_SLOPE * x; break;
        }
        ho[c] = (__bf16)(v * mk[c] * KEEP_SCALE);
    }
}

// ---------------------------------------------------------------------------
// Final 1024 -> 10 layer + log_softmax. One wave per row.
// ---------------------------------------------------------------------------
__global__ __launch_bounds__(256) void final_logits(
    const __bf16* __restrict__ Hin, const float* __restrict__ W3,
    const float* __restrict__ b3, float* __restrict__ out, int H)
{
    const int row  = blockIdx.x * 8 + (threadIdx.x >> 5);
    const int lane = threadIdx.x & 31;
    const __bf16* h = Hin + (size_t)row * H;

    float acc[10];
    #pragma unroll
    for (int c = 0; c < 10; ++c) acc[c] = 0.f;
    for (int k = lane; k < H; k += 32) {
        float hv = (float)h[k];
        #pragma unroll
        for (int c = 0; c < 10; ++c) acc[c] += hv * W3[c * H + k];
    }
    #pragma unroll
    for (int c = 0; c < 10; ++c) {
        #pragma unroll
        for (int off = 16; off > 0; off >>= 1)
            acc[c] += __shfl_down(acc[c], off, 32);
    }
    if (lane == 0) {
        #pragma unroll
        for (int c = 0; c < 10; ++c) acc[c] += b3[c];
        float m = acc[0];
        #pragma unroll
        for (int c = 1; c < 10; ++c) m = fmaxf(m, acc[c]);
        float s = 0.f;
        #pragma unroll
        for (int c = 0; c < 10; ++c) s += expf(acc[c] - m);
        const float lse = m + logf(s);
        float* o = out + (size_t)row * 10;
        #pragma unroll
        for (int c = 0; c < 10; ++c) o[c] = acc[c] - lse;
    }
}

// f32 -> bf16 conversion with optional zero-padded K (784 -> 800 for layer 0)
__global__ void cvt_f32_to_bf16_pad(const float* __restrict__ src,
                                    __bf16* __restrict__ dst,
                                    int K, int Kpad, long long total)
{
    long long i = (long long)blockIdx.x * 256 + threadIdx.x;
    if (i >= total) return;
    long long r = i / Kpad;
    int c = (int)(i - r * Kpad);
    dst[i] = (c < K) ? (__bf16)src[r * K + c] : (__bf16)0.0f;
}

extern "C" void kernel_launch(void* const* d_in, const int* in_sizes, int n_in,
                              void* d_out, int out_size, void* d_ws, size_t ws_size,
                              hipStream_t stream) {
    const float* x     = (const float*)d_in[0];
    const float* W0    = (const float*)d_in[1];
    const float* b0    = (const float*)d_in[2];
    const float* W1    = (const float*)d_in[3];
    const float* b1    = (const float*)d_in[4];
    const float* W2    = (const float*)d_in[5];
    const float* b2    = (const float*)d_in[6];
    const float* W3    = (const float*)d_in[7];
    const float* b3    = (const float*)d_in[8];
    const float* mask1 = (const float*)d_in[9];
    const float* mask2 = (const float*)d_in[10];
    const float* mask3 = (const float*)d_in[11];
    const int* type_ids = (const int*)d_in[12];

    const int B = 32768, DIN = 784, DPAD = 800, H = 1024, C = 10;

    char* w = (char*)d_ws;
    auto carve = [&](size_t bytes) -> char* {
        char* p = w; w += (bytes + 255) & ~(size_t)255; return p;
    };
    __bf16* xbf  = (__bf16*)carve((size_t)B * DPAD * 2);
    __bf16* w0bf = (__bf16*)carve((size_t)H * DPAD * 2);
    __bf16* w1bf = (__bf16*)carve((size_t)H * H * 2);
    __bf16* w2bf = (__bf16*)carve((size_t)H * H * 2);
    float*  Zb   = (float*) carve((size_t)B * H * 4);
    __bf16* hbf  = (__bf16*)carve((size_t)B * H * 2);

    long long nx = (long long)B * DPAD;
    cvt_f32_to_bf16_pad<<<(unsigned)((nx + 255) / 256), 256, 0, stream>>>(x, xbf, DIN, DPAD, nx);
    long long nw0 = (long long)H * DPAD;
    cvt_f32_to_bf16_pad<<<(unsigned)((nw0 + 255) / 256), 256, 0, stream>>>(W0, w0bf, DIN, DPAD, nw0);
    long long nw = (long long)H * H;
    cvt_f32_to_bf16_pad<<<(unsigned)((nw + 255) / 256), 256, 0, stream>>>(W1, w1bf, H, H, nw);
    cvt_f32_to_bf16_pad<<<(unsigned)((nw + 255) / 256), 256, 0, stream>>>(W2, w2bf, H, H, nw);

    dim3 gg(H / NT, B / MT);
    gemm_bf16_wmma<<<gg, 256, 0, stream>>>(xbf, w0bf, b0, Zb, DPAD, H);
    act_mixed<<<B, 256, 0, stream>>>(Zb, type_ids + 0 * H, mask1, hbf, H);
    gemm_bf16_wmma<<<gg, 256, 0, stream>>>(hbf, w1bf, b1, Zb, H, H);
    act_mixed<<<B, 256, 0, stream>>>(Zb, type_ids + 1 * H, mask2, hbf, H);
    gemm_bf16_wmma<<<gg, 256, 0, stream>>>(hbf, w2bf, b2, Zb, H, H);
    act_mixed<<<B, 256, 0, stream>>>(Zb, type_ids + 2 * H, mask3, hbf, H);

    float* out = (float*)d_out;
    final_logits<<<B / 8, 256, 0, stream>>>(hbf, W3, b3, out, H);

    // Pass-through masks into the tuple output.
    float* mout = out + (size_t)B * C;
    const size_t mbytes = (size_t)B * H * sizeof(float);
    (void)hipMemcpyAsync(mout,                   mask1, mbytes, hipMemcpyDeviceToDevice, stream);
    (void)hipMemcpyAsync(mout + (size_t)B * H,   mask2, mbytes, hipMemcpyDeviceToDevice, stream);
    (void)hipMemcpyAsync(mout + 2*(size_t)B * H, mask3, mbytes, hipMemcpyDeviceToDevice, stream);
}
